// LocalEnergyCore_52467320488438
// MI455X (gfx1250) — matrix operand order repaired and checked by reading, not compiled, not run on hardware
//
#include <hip/hip_runtime.h>
#include <hip/hip_bf16.h>

typedef __attribute__((ext_vector_type(16))) _Float16   v16h;
typedef __attribute__((ext_vector_type(8)))  float      v8f;
typedef __attribute__((ext_vector_type(4)))  unsigned   u32x4;

#define Bsz    256
#define Kch    16
#define Hh     64
#define Ww     64
#define Nsites 1024
#define FKEEP  143
#define FPAD   160     // 143 padded up to multiple of 32 for the WMMA K loop
#define HID    128
#define NCLS   16
#define MC     64      // batch rows per chunk (4 chunks of 64 = 256)

// d_ws layout (half units unless noted)
#define W1T_OFF 0                              // half[HID][FPAD]  (col-major W1, K-padded)
#define W2T_OFF (HID * FPAD)                   // half[NCLS][HID]  (col-major W2)
#define CNT_OFF_BYTES ((HID * FPAD + NCLS * HID) * 2)

// ---------------------------------------------------------------------------
// Prep: transpose/convert weights to f16 fragment-friendly layout; zero counter
// ---------------------------------------------------------------------------
__global__ void prep_kernel(const float* __restrict__ W1, const float* __restrict__ W2,
                            _Float16* __restrict__ wsH, unsigned int* __restrict__ cnt) {
    int i = blockIdx.x * blockDim.x + threadIdx.x;
    if (i == 0) *cnt = 0u;
    if (i < HID * FPAD) {
        int n = i / FPAD, k = i % FPAD;
        float v = (k < FKEEP) ? W1[k * HID + n] : 0.0f;   // W1 is (143,128) row-major
        wsH[W1T_OFF + i] = (_Float16)v;
    } else if (i < HID * FPAD + NCLS * HID) {
        int j = i - HID * FPAD;
        int c = j / HID, k = j % HID;
        wsH[W2T_OFF + j] = (_Float16)W2[k * NCLS + c];    // W2 is (128,16) row-major
    }
}

// ---------------------------------------------------------------------------
// Main: one workgroup per site n; 512 threads = 16 wave32s
// ---------------------------------------------------------------------------
__global__ __launch_bounds__(512)
void site_kernel(const int* __restrict__ z, const int* __restrict__ bs,
                 const int* __restrict__ ii, const int* __restrict__ jj,
                 const float* __restrict__ b1, const float* __restrict__ b2,
                 const _Float16* __restrict__ wsH, unsigned int* __restrict__ cnt) {
    __shared__ _Float16 ctx[MC * FPAD];   // 20 KB: f16 context tile (drop applied, K-padded)
    __shared__ _Float16 hb[MC * HID];     // 16 KB: relu(hidden) as f16
    __shared__ float    lg[MC * NCLS];    //  4 KB: logits f32
    __shared__ unsigned int s_err;

    const int site = blockIdx.x;
    const int tid  = threadIdx.x;
    const int lane = tid & 31;
    const int wv   = tid >> 5;            // 0..15
    const int bsite = bs[site];
    const int is    = ii[site];
    const int js    = jj[site];
    const int drop  = 4 * Kch + bsite;    // center cell, channel bs

    if (tid == 0) s_err = 0u;
    __syncthreads();

    for (int ch = 0; ch < Bsz / MC; ++ch) {
        const int b0 = ch * MC;

        // ---- zero K-pad columns 143..159 -------------------------------
        for (int t = tid; t < MC * (FPAD - FKEEP); t += blockDim.x) {
            int r = t / (FPAD - FKEEP);
            int c = FKEEP + t % (FPAD - FKEEP);
            ctx[r * FPAD + c] = (_Float16)0.0f;
        }
        // ---- gather 3x3xK context, apply feature drop ------------------
        for (int t = tid; t < MC * 9 * Kch; t += blockDim.x) {
            int r = t / (9 * Kch);
            int g = t % (9 * Kch);              // feature = pos*16 + channel
            int p = g >> 4, k = g & 15;
            int ni = (is + (p / 3) + Hh - 1) & (Hh - 1);
            int nj = (js + (p % 3) + Ww - 1) & (Ww - 1);
            int b  = b0 + r;
            int zv = z[((b * Kch + k) * Hh + ni) * Ww + nj];
            if (g != drop) {
                int col = (g < drop) ? g : g - 1;
                ctx[r * FPAD + col] = (_Float16)((float)zv * (1.0f / 15.0f));
            }
        }
        __syncthreads();

        // ---- GEMM1: (64x160) @ (160x128) -> relu -> hb (f16) -----------
        // 4 M-tiles x 8 N-tiles over 16 waves: wave w -> mt = w>>2, nt = (w&3)*2 + {0,1}
        {
            const int mt  = wv >> 2;
            const int ntb = (wv & 3) * 2;
            const int rlo = lane & 15;
            const int hi  = lane >> 4;
            v8f acc0 = {}; v8f acc1 = {};
            #pragma unroll
            for (int kt = 0; kt < FPAD / 32; ++kt) {
                const int kb = kt * 32;
                union { v16h h; u32x4 q[2]; } A, B0, B1;
                // A fragment: lanes 0-15 halves kb..kb+15; hi lanes +16
                const u32x4* pa = (const u32x4*)&ctx[(mt * 16 + rlo) * FPAD + kb + hi * 16];
                A.q[0] = pa[0];
                A.q[1] = pa[1];
                // B fragment: lane = column; 16 contiguous halves (half-wave K split)
                const u32x4* pb0 = (const u32x4*)&wsH[W1T_OFF + (ntb * 16 + rlo) * FPAD + kb + hi * 16];
                B0.q[0] = pb0[0]; B0.q[1] = pb0[1];
                const u32x4* pb1 = (const u32x4*)&wsH[W1T_OFF + ((ntb + 1) * 16 + rlo) * FPAD + kb + hi * 16];
                B1.q[0] = pb1[0]; B1.q[1] = pb1[1];
                acc0 = __builtin_amdgcn_wmma_f32_16x16x32_f16(false, A.h, false, B0.h,
                                                              (short)0, acc0, false, false);
                acc1 = __builtin_amdgcn_wmma_f32_16x16x32_f16(false, A.h, false, B1.h,
                                                              (short)0, acc1, false, false);
            }
            const int col0 = ntb * 16 + rlo;
            const int col1 = col0 + 16;
            const float bb0 = b1[col0];
            const float bb1 = b1[col1];
            #pragma unroll
            for (int r = 0; r < 8; ++r) {                 // D layout: row = r + 8*hi
                int M = mt * 16 + r + hi * 8;
                float v0 = acc0[r] + bb0; v0 = v0 > 0.0f ? v0 : 0.0f;
                float v1 = acc1[r] + bb1; v1 = v1 > 0.0f ? v1 : 0.0f;
                hb[M * HID + col0] = (_Float16)v0;
                hb[M * HID + col1] = (_Float16)v1;
            }
        }
        __syncthreads();

        // ---- GEMM2: (64x128) @ (128x16) -> logits (f32) ----------------
        if (wv < 4) {                                     // wave-uniform branch (EXEC all-1s)
            const int mt  = wv;
            const int rlo = lane & 15;
            const int hi  = lane >> 4;
            v8f acc = {};
            #pragma unroll
            for (int kt = 0; kt < HID / 32; ++kt) {
                const int kb = kt * 32;
                union { v16h h; u32x4 q[2]; } A, Bf;
                const u32x4* pa = (const u32x4*)&hb[(mt * 16 + rlo) * HID + kb + hi * 16];
                A.q[0] = pa[0];
                A.q[1] = pa[1];
                const u32x4* pb = (const u32x4*)&wsH[W2T_OFF + rlo * HID + kb + hi * 16];
                Bf.q[0] = pb[0]; Bf.q[1] = pb[1];
                acc = __builtin_amdgcn_wmma_f32_16x16x32_f16(false, A.h, false, Bf.h,
                                                             (short)0, acc, false, false);
            }
            const float bb = b2[rlo];
            #pragma unroll
            for (int r = 0; r < 8; ++r) {
                int M = mt * 16 + r + hi * 8;
                lg[M * NCLS + rlo] = acc[r] + bb;
            }
        }
        __syncthreads();

        // ---- argmax + compare vs target --------------------------------
        if (tid < MC) {
            const float* row = &lg[tid * NCLS];
            float best = row[0]; int bi = 0;
            #pragma unroll
            for (int c = 1; c < NCLS; ++c) {
                float v = row[c];
                if (v > best) { best = v; bi = c; }       // strict > == first-max (jnp.argmax)
            }
            int b   = b0 + tid;
            int tgt = z[((b * Kch + bsite) * Hh + is) * Ww + js];
            if (bi != tgt) atomicAdd(&s_err, 1u);
        }
        __syncthreads();                                  // ctx reused next chunk
    }
    if (tid == 0 && s_err) atomicAdd(cnt, s_err);
}

// ---------------------------------------------------------------------------
__global__ void finalize_kernel(const unsigned int* __restrict__ cnt, float* __restrict__ out) {
    out[0] = (float)(*cnt) * (1.0f / (float)(Nsites * Bsz));
}

// ---------------------------------------------------------------------------
extern "C" void kernel_launch(void* const* d_in, const int* in_sizes, int n_in,
                              void* d_out, int out_size, void* d_ws, size_t ws_size,
                              hipStream_t stream) {
    const int*   z   = (const int*)d_in[0];
    const int*   bsp = (const int*)d_in[1];
    const int*   iip = (const int*)d_in[2];
    const int*   jjp = (const int*)d_in[3];
    const float* W1  = (const float*)d_in[4];
    const float* b1  = (const float*)d_in[5];
    const float* W2  = (const float*)d_in[6];
    const float* b2  = (const float*)d_in[7];

    _Float16*     wsH = (_Float16*)d_ws;
    unsigned int* cnt = (unsigned int*)((char*)d_ws + CNT_OFF_BYTES);

    const int prepElems = HID * FPAD + NCLS * HID;        // 22528
    prep_kernel<<<(prepElems + 255) / 256, 256, 0, stream>>>(W1, W2, wsH, cnt);
    site_kernel<<<Nsites, 512, 0, stream>>>(z, bsp, iip, jjp, b1, b2, wsH, cnt);
    finalize_kernel<<<1, 1, 0, stream>>>(cnt, (float*)d_out);
}
